// Transformer_76819785056415
// MI455X (gfx1250) — matrix operand order repaired and checked by reading, not compiled, not run on hardware
//
#include <hip/hip_runtime.h>

// ---------------------------------------------------------------------------
// Model constants
// ---------------------------------------------------------------------------
#define BB 2
#define TT 2048
#define DD 1024
#define HH 16
#define KHH 4
#define HDD 64
#define FF 4096
#define LL 4
#define VV 32000
#define MM (BB * TT)          // 4096 token rows
#define SCALE 0.125f          // HD^-0.5
#define EPSV 1e-6f

typedef __attribute__((ext_vector_type(16))) __bf16 v16bf;
typedef __attribute__((ext_vector_type(8)))  float  v8f;
typedef unsigned short ushort_t;

// ---- gfx1250 async-to-LDS support (guarded; falls back to uint4 copies) ----
#if defined(__has_builtin)
#if __has_builtin(__builtin_amdgcn_global_load_async_to_lds_b128) && \
    __has_builtin(__builtin_amdgcn_s_wait_asynccnt)
#define HAVE_ASYNC 1
#endif
#endif
#ifndef HAVE_ASYNC
#define HAVE_ASYNC 0
#endif

#if HAVE_ASYNC
typedef int int4v __attribute__((ext_vector_type(4)));
typedef __attribute__((address_space(1))) int4v as1_int4;
typedef __attribute__((address_space(3))) int4v as3_int4;
__device__ __forceinline__ void async_ld16(void* lds, const void* g) {
    // Generic LDS pointers keep the LDS byte offset in the low 32 bits
    // (ISA 10.2); global pointers are identical in AS1.
    __builtin_amdgcn_global_load_async_to_lds_b128(
        (as1_int4*)(unsigned long long)(size_t)g,
        (as3_int4*)(unsigned int)(size_t)lds,
        0, 0);
}
#define ASYNC_WAIT() __builtin_amdgcn_s_wait_asynccnt(0)
#else
__device__ __forceinline__ void async_ld16(void* lds, const void* g) {
    *(uint4*)lds = *(const uint4*)g;
}
#define ASYNC_WAIT() ((void)0)
#endif

union FragBF {
    v16bf v;
    unsigned int u[8];
};

__device__ __forceinline__ ushort_t f2bf(float f) {
    unsigned int u = __float_as_uint(f);
    u += 0x7FFFu + ((u >> 16) & 1u);       // round-to-nearest-even
    return (ushort_t)(u >> 16);
}

__device__ __forceinline__ v8f wmma_bf16(const FragBF& a, const FragBF& b, v8f c) {
    return __builtin_amdgcn_wmma_f32_16x16x32_bf16(
        false, a.v, false, b.v, (short)0, c, false, false);
}

// A-fragment (16x32 bf16, row-major source, row stride ld elements).
// Lane L (r=L&15, hi=L>>4) holds row r:
//   v0..7 : K = 2v + hi*16
__device__ __forceinline__ void load_a_frag(FragBF& f, const ushort_t* __restrict__ src,
                                            int ld, int lane) {
    int r = lane & 15, hi = lane >> 4;
    const ushort_t* p = src + (size_t)r * ld + hi * 16;
#pragma unroll
    for (int v = 0; v < 8; ++v) f.u[v]     = *(const unsigned int*)(p + 2 * v);
}

// B-fragment (32x16 bf16) from k-major source: element (k,n) at src[k*ld+n].
__device__ __forceinline__ void load_b_frag_kmajor(FragBF& f, const ushort_t* __restrict__ src,
                                                   int ld, int lane) {
    int n = lane & 15, hi = lane >> 4;
    const ushort_t* p = src + n + (size_t)(hi * 16) * ld;
#pragma unroll
    for (int v = 0; v < 8; ++v) {
        unsigned int lo = p[(size_t)(2 * v) * ld];
        unsigned int hh = p[(size_t)(2 * v + 1) * ld];
        f.u[v] = lo | (hh << 16);
    }
}

// B-fragment from n-major (transposed) source: element (k,n) at src[n*ld+k].
__device__ __forceinline__ void load_b_frag_nmajor(FragBF& f, const ushort_t* __restrict__ src,
                                                   int ld, int lane) {
    int n = lane & 15, hi = lane >> 4;
    const ushort_t* p = src + (size_t)n * ld + hi * 16;
#pragma unroll
    for (int v = 0; v < 8; ++v) f.u[v] = *(const unsigned int*)(p + 2 * v);
}

// ---------------------------------------------------------------------------
// Embedding: x[m,:] = embed_w[ids[m],:] + pos_w[m % T, :]   (float4 vectorized)
// ---------------------------------------------------------------------------
__global__ __launch_bounds__(256)
void embed_kernel(const int* __restrict__ ids, const float* __restrict__ E,
                  const float* __restrict__ P, float* __restrict__ X) {
    int m = blockIdx.x;
    int t = m & (TT - 1);
    size_t id = (size_t)ids[m];
    const float4* e4 = (const float4*)(E + id * DD);
    const float4* p4 = (const float4*)(P + (size_t)t * DD);
    float4* x4 = (float4*)(X + (size_t)m * DD);
    float4 a = e4[threadIdx.x], b = p4[threadIdx.x];
    x4[threadIdx.x] = make_float4(a.x + b.x, a.y + b.y, a.z + b.z, a.w + b.w);
}

// ---------------------------------------------------------------------------
// fp32 -> bf16 bulk convert, 4 elements per thread (grid-stride)
// ---------------------------------------------------------------------------
__global__ __launch_bounds__(256)
void f32_to_bf16_kernel(const float* __restrict__ in, ushort_t* __restrict__ out,
                        long long n4) {
    long long i = (long long)blockIdx.x * blockDim.x + threadIdx.x;
    long long stride = (long long)gridDim.x * blockDim.x;
    const float4* in4 = (const float4*)in;
    uint2* out2 = (uint2*)out;
    for (; i < n4; i += stride) {
        float4 v = in4[i];
        uint2 o;
        o.x = (unsigned int)f2bf(v.x) | ((unsigned int)f2bf(v.y) << 16);
        o.y = (unsigned int)f2bf(v.z) | ((unsigned int)f2bf(v.w) << 16);
        out2[i] = o;
    }
}

// ---------------------------------------------------------------------------
// RMSNorm: one block per row (D = 1024), bf16 output, float4 vectorized
// ---------------------------------------------------------------------------
__global__ __launch_bounds__(256)
void rmsnorm_kernel(const float* __restrict__ X, const float* __restrict__ Wn,
                    ushort_t* __restrict__ Out) {
    int row = blockIdx.x;
    int tid = threadIdx.x, wid = tid >> 5, lane = tid & 31;
    const float4* x4 = (const float4*)(X + (size_t)row * DD);
    float4 v = x4[tid];
    float s = v.x * v.x + v.y * v.y + v.z * v.z + v.w * v.w;
#pragma unroll
    for (int m = 1; m < 32; m <<= 1) s += __shfl_xor(s, m, 32);
    __shared__ float red[8];
    if (lane == 0) red[wid] = s;
    __syncthreads();
    float tot = 0.f;
#pragma unroll
    for (int w = 0; w < 8; ++w) tot += red[w];
    float r = rsqrtf(tot * (1.0f / DD) + EPSV);
    float4 wv = ((const float4*)Wn)[tid];
    uint2 o;
    o.x = (unsigned int)f2bf(v.x * r * wv.x) | ((unsigned int)f2bf(v.y * r * wv.y) << 16);
    o.y = (unsigned int)f2bf(v.z * r * wv.z) | ((unsigned int)f2bf(v.w * r * wv.w) << 16);
    ((uint2*)(Out + (size_t)row * DD))[tid] = o;
}

// ---------------------------------------------------------------------------
// WMMA GEMM:  C[M,N] = A_bf16[M,K] @ B_bf16[K,N]   (BT: B stored [N,K])
// Block tile 64x256, BK=64, 256 threads = 8 waves (2x4), wave tile 32x64.
// 16 WMMAs per wave per barrier pair; async-to-LDS staging on gfx1250.
// Epilogues: 0 = store f32, 1 = residual-add f32, 2 = store bf16, 3 = GELU->bf16
// ---------------------------------------------------------------------------
template <int EPI, bool BT>
__global__ __launch_bounds__(256)
void gemm_kernel(const ushort_t* __restrict__ A, const ushort_t* __restrict__ Bm,
                 const float* __restrict__ Res, void* __restrict__ OutP,
                 int M, int N, int K) {
    __shared__ ushort_t As[64 * 64];      //  8 KB
    __shared__ ushort_t Bs[64 * 256];     // 32 KB

    int tid = threadIdx.x, wid = tid >> 5, lane = tid & 31;
    int wm = wid >> 2, wn = wid & 3;      // wave tile: rows wm*32+.., cols wn*64+..
    int m0 = blockIdx.y * 64, n0 = blockIdx.x * 256;

    v8f acc[2][4] = {};

    for (int k0 = 0; k0 < K; k0 += 64) {
        // ---- stage A tile (64x64): 2 x 16B per thread
#pragma unroll
        for (int i = 0; i < 2; ++i) {
            int idx = (tid + i * 256) * 8;
            int r = idx >> 6, c = idx & 63;
            async_ld16(&As[r * 64 + c], &A[(size_t)(m0 + r) * K + k0 + c]);
        }
        // ---- stage B tile (64x256, k-major in LDS)
        if (!BT) {
#pragma unroll
            for (int i = 0; i < 8; ++i) {
                int idx = (tid + i * 256) * 8;
                int r = idx >> 8, c = idx & 255;
                async_ld16(&Bs[r * 256 + c], &Bm[(size_t)(k0 + r) * N + n0 + c]);
            }
        } else {
#pragma unroll
            for (int i = 0; i < 8; ++i) {
                int idx = (tid + i * 256) * 8;
                int n = idx >> 6, kk = idx & 63;
                ushort_t tmp[8];
                *(uint4*)tmp = *(const uint4*)&Bm[(size_t)(n0 + n) * K + k0 + kk];
#pragma unroll
                for (int j = 0; j < 8; ++j) Bs[(kk + j) * 256 + n] = tmp[j];
            }
        }

        // prefetch next K-tile while the WMMA pipeline drains this one
        if (k0 + 64 < K) {
            int idx = tid * 8;
            __builtin_prefetch(&A[(size_t)(m0 + (idx >> 6)) * K + k0 + 64 + (idx & 63)], 0, 1);
            if (!BT) {
                __builtin_prefetch(&Bm[(size_t)(k0 + 64 + (idx >> 8)) * N + n0 + (idx & 255)], 0, 1);
            }
        }

        ASYNC_WAIT();
        __syncthreads();

        // ---- two K=32 steps over the staged tile
#pragma unroll
        for (int ks = 0; ks < 2; ++ks) {
            FragBF a0, a1;
            load_a_frag(a0, As + (wm * 32) * 64 + ks * 32, 64, lane);
            load_a_frag(a1, As + (wm * 32 + 16) * 64 + ks * 32, 64, lane);
#pragma unroll
            for (int nf = 0; nf < 4; ++nf) {
                FragBF bfr;
                load_b_frag_kmajor(bfr, Bs + (ks * 32) * 256 + wn * 64 + nf * 16, 256, lane);
                acc[0][nf] = wmma_bf16(a0, bfr, acc[0][nf]);
                acc[1][nf] = wmma_bf16(a1, bfr, acc[1][nf]);
            }
        }
        __syncthreads();
    }

    // ---- epilogue: C layout: VGPR v <-> row v + 8*hi, col = lane&15
    int hi = lane >> 4, col = lane & 15;
    float* outf = (float*)OutP;
    ushort_t* outb = (ushort_t*)OutP;
#pragma unroll
    for (int fx = 0; fx < 2; ++fx) {
#pragma unroll
        for (int nf = 0; nf < 4; ++nf) {
            int rb = m0 + wm * 32 + fx * 16;
            int cb = n0 + wn * 64 + nf * 16;
#pragma unroll
            for (int v = 0; v < 8; ++v) {
                size_t R = (size_t)(rb + v + 8 * hi);
                size_t idx = R * (size_t)N + cb + col;
                float val = acc[fx][nf][v];
                if (EPI == 0) {
                    outf[idx] = val;
                } else if (EPI == 1) {
                    outf[idx] = Res[idx] + val;
                } else if (EPI == 2) {
                    outb[idx] = f2bf(val);
                } else {  // exact GELU
                    float g = 0.5f * val * (1.0f + erff(val * 0.70710678118654752f));
                    outb[idx] = f2bf(g);
                }
            }
        }
    }
}

// ---------------------------------------------------------------------------
// Causal GQA flash attention.
// One wave per 16-query tile per (b,h); 4 waves / block.
// K/V chunks staged per-wave into LDS with async 16B transfers; fragments
// built from LDS. P round-trips through LDS to reach A-fragment layout.
// ---------------------------------------------------------------------------
__global__ __launch_bounds__(128)
void attn_kernel(const ushort_t* __restrict__ Q, const ushort_t* __restrict__ KV,
                 ushort_t* __restrict__ O) {
    __shared__ ushort_t Ks[4][32 * 64];
    __shared__ ushort_t Vs[4][32 * 64];
    __shared__ ushort_t Ps[4][16 * 32];

    int tid = threadIdx.x, wid = tid >> 5, lane = tid & 31;
    int W = blockIdx.x * 4 + wid;
    int qt = W & 127;             // T/16 = 128 query tiles
    int h  = (W >> 7) & 15;
    int b  = W >> 11;
    int q0 = qt * 16;
    int kvh = h >> 2;             // GQA: G = 4
    int hi = lane >> 4, col = lane & 15;

    // Q fragments (16 rows x 64 dims -> two K=32 A-frags), loaded once
    const ushort_t* qptr = Q + ((size_t)(b * TT + q0)) * (HH * HDD) + h * HDD;
    FragBF qa0, qa1;
    load_a_frag(qa0, qptr, HH * HDD, lane);
    load_a_frag(qa1, qptr + 32, HH * HDD, lane);

    v8f acc0 = {}, acc1 = {}, acc2 = {}, acc3 = {};
    float mrow[8], lrow[8];
#pragma unroll
    for (int v = 0; v < 8; ++v) { mrow[v] = -1e30f; lrow[v] = 0.f; }

    ushort_t* ksw = &Ks[wid][0];
    ushort_t* vsw = &Vs[wid][0];

    int jmax = (q0 + 15) >> 5;
    for (int j = 0; j <= jmax; ++j) {
        int kb = j * 32;
        const ushort_t* kvbase = KV + (size_t)(b * TT + kb) * 512 + kvh * HDD;

        // ---- stage K/V chunk (32 keys x 64 dims each) with 16B transfers.
        // lane owns key row `lane`: 8 x 16B for K, 8 x 16B for V.
#pragma unroll
        for (int i = 0; i < 8; ++i) {
            async_ld16(ksw + lane * 64 + i * 8, kvbase + (size_t)lane * 512 + i * 8);
            async_ld16(vsw + lane * 64 + i * 8, kvbase + 256 + (size_t)lane * 512 + i * 8);
        }
        ASYNC_WAIT();   // per-wave region: no cross-wave barrier needed

        // ---- S = Q @ K^T : two 16-key halves, HD=64 split into two K=32 steps
        FragBF kf;
        v8f s0 = {}, s1 = {};
        load_b_frag_nmajor(kf, ksw, 64, lane);            s0 = wmma_bf16(qa0, kf, s0);
        load_b_frag_nmajor(kf, ksw + 32, 64, lane);       s0 = wmma_bf16(qa1, kf, s0);
        load_b_frag_nmajor(kf, ksw + 16 * 64, 64, lane);  s1 = wmma_bf16(qa0, kf, s1);
        load_b_frag_nmajor(kf, ksw + 16 * 64 + 32, 64, lane); s1 = wmma_bf16(qa1, kf, s1);

        // ---- scale + causal mask + online softmax (half-wave row reductions)
#pragma unroll
        for (int v = 0; v < 8; ++v) {
            int qg = q0 + v + 8 * hi;
            float x0 = s0[v] * SCALE;
            float x1 = s1[v] * SCALE;
            if (kb + col > qg)      x0 = -1e30f;
            if (kb + 16 + col > qg) x1 = -1e30f;
            float mx = fmaxf(x0, x1);
#pragma unroll
            for (int m = 1; m < 16; m <<= 1) mx = fmaxf(mx, __shfl_xor(mx, m, 32));
            float mnew = fmaxf(mrow[v], mx);
            float alpha = __expf(mrow[v] - mnew);
            float p0 = __expf(x0 - mnew);
            float p1 = __expf(x1 - mnew);
            float rs = p0 + p1;
#pragma unroll
            for (int m = 1; m < 16; m <<= 1) rs += __shfl_xor(rs, m, 32);
            lrow[v] = lrow[v] * alpha + rs;
            mrow[v] = mnew;
            acc0[v] *= alpha; acc1[v] *= alpha; acc2[v] *= alpha; acc3[v] *= alpha;
            int pr = v + 8 * hi;
            Ps[wid][pr * 32 + col]      = f2bf(p0);
            Ps[wid][pr * 32 + 16 + col] = f2bf(p1);
        }

        // ---- P (16x32) out of LDS in A-fragment layout, then O += P @ V
        FragBF pa, vf;
        load_a_frag(pa, Ps[wid], 32, lane);
        load_b_frag_kmajor(vf, vsw, 64, lane);       acc0 = wmma_bf16(pa, vf, acc0);
        load_b_frag_kmajor(vf, vsw + 16, 64, lane);  acc1 = wmma_bf16(pa, vf, acc1);
        load_b_frag_kmajor(vf, vsw + 32, 64, lane);  acc2 = wmma_bf16(pa, vf, acc2);
        load_b_frag_kmajor(vf, vsw + 48, 64, lane);  acc3 = wmma_bf16(pa, vf, acc3);
    }

#pragma unroll
    for (int v = 0; v < 8; ++v) {
        float inv = 1.0f / lrow[v];
        size_t orow = (size_t)(b * TT + q0 + v + 8 * hi) * (HH * HDD) + h * HDD + col;
        O[orow + 0]  = f2bf(acc0[v] * inv);
        O[orow + 16] = f2bf(acc1[v] * inv);
        O[orow + 32] = f2bf(acc2[v] * inv);
        O[orow + 48] = f2bf(acc3[v] * inv);
    }
}

// ---------------------------------------------------------------------------
// Host orchestration
// ---------------------------------------------------------------------------
extern "C" void kernel_launch(void* const* d_in, const int* in_sizes, int n_in,
                              void* d_out, int out_size, void* d_ws, size_t ws_size,
                              hipStream_t stream) {
    const int*   ids     = (const int*)d_in[0];
    const float* embed_w = (const float*)d_in[1];
    const float* pos_w   = (const float*)d_in[2];
    const float* ln1_w   = (const float*)d_in[3];
    const float* ln2_w   = (const float*)d_in[4];
    const float* wq      = (const float*)d_in[5];
    const float* wkv     = (const float*)d_in[6];
    const float* wo      = (const float*)d_in[7];
    const float* w1      = (const float*)d_in[8];
    const float* w2      = (const float*)d_in[9];
    const float* norm_w  = (const float*)d_in[10];

    // ---- workspace carve-up
    char* ws = (char*)d_ws;
    float*    x    = (float*)ws;       ws += (size_t)MM * DD * 4;      // residual fp32
    ushort_t* hbuf = (ushort_t*)ws;    ws += (size_t)MM * DD * 2;      // normed acts bf16
    ushort_t* qb   = (ushort_t*)ws;    ws += (size_t)MM * (HH*HDD) * 2;
    ushort_t* kvb  = (ushort_t*)ws;    ws += (size_t)MM * 512 * 2;
    ushort_t* ob   = (ushort_t*)ws;    ws += (size_t)MM * (HH*HDD) * 2;
    ushort_t* gb   = (ushort_t*)ws;    ws += (size_t)MM * FF * 2;      // MLP hidden bf16
    ushort_t* wqb  = (ushort_t*)ws;    ws += (size_t)DD * (HH*HDD) * 2;
    ushort_t* wkvb = (ushort_t*)ws;    ws += (size_t)DD * 512 * 2;
    ushort_t* wob  = (ushort_t*)ws;    ws += (size_t)(HH*HDD) * DD * 2;
    ushort_t* w1b  = (ushort_t*)ws;    ws += (size_t)DD * FF * 2;
    ushort_t* w2b  = (ushort_t*)ws;    ws += (size_t)FF * DD * 2;
    ushort_t* embb = (ushort_t*)ws;    ws += (size_t)VV * DD * 2;

    auto conv = [&](const float* src, ushort_t* dst, long long n) {
        long long n4 = n >> 2;
        int blocks = (int)((n4 + 256LL * 4 - 1) / (256LL * 4));
        if (blocks > 8192) blocks = 8192;
        f32_to_bf16_kernel<<<blocks, 256, 0, stream>>>(src, dst, n4);
    };

    // x = embed[ids] + pos
    embed_kernel<<<MM, 256, 0, stream>>>(ids, embed_w, pos_w, x);

    for (int l = 0; l < LL; ++l) {
        // per-layer weight conversion to bf16 staging
        conv(wq  + (size_t)l * DD * (HH*HDD), wqb,  (long long)DD * (HH*HDD));
        conv(wkv + (size_t)l * DD * 512,      wkvb, (long long)DD * 512);
        conv(wo  + (size_t)l * (HH*HDD) * DD, wob,  (long long)(HH*HDD) * DD);
        conv(w1  + (size_t)l * DD * FF,       w1b,  (long long)DD * FF);
        conv(w2  + (size_t)l * FF * DD,       w2b,  (long long)FF * DD);

        // h = rmsnorm(x, ln1)
        rmsnorm_kernel<<<MM, 256, 0, stream>>>(x, ln1_w + (size_t)l * DD, hbuf);

        // q = h @ wq (bf16 out) ; kv = h @ wkv (bf16 out)
        gemm_kernel<2, false><<<dim3(1024 / 256, MM / 64), 256, 0, stream>>>(
            hbuf, wqb, nullptr, qb, MM, 1024, DD);
        gemm_kernel<2, false><<<dim3(512 / 256, MM / 64), 256, 0, stream>>>(
            hbuf, wkvb, nullptr, kvb, MM, 512, DD);

        // o = softmax(q k^T) v  (flash, causal, GQA)
        attn_kernel<<<(BB * HH * (TT / 16)) / 4, 128, 0, stream>>>(qb, kvb, ob);

        // x += o @ wo
        gemm_kernel<1, false><<<dim3(DD / 256, MM / 64), 256, 0, stream>>>(
            ob, wob, x, x, MM, DD, HH * HDD);

        // h = rmsnorm(x, ln2) ; g = gelu(h @ w1) ; x += g @ w2
        rmsnorm_kernel<<<MM, 256, 0, stream>>>(x, ln2_w + (size_t)l * DD, hbuf);
        gemm_kernel<3, false><<<dim3(FF / 256, MM / 64), 256, 0, stream>>>(
            hbuf, w1b, nullptr, gb, MM, FF, DD);
        gemm_kernel<1, false><<<dim3(DD / 256, MM / 64), 256, 0, stream>>>(
            gb, w2b, x, x, MM, DD, FF);
    }

    // tied lm-head: logits = rmsnorm(x, norm_w) @ embed^T
    conv(embed_w, embb, (long long)VV * DD);
    rmsnorm_kernel<<<MM, 256, 0, stream>>>(x, norm_w, hbuf);
    gemm_kernel<0, true><<<dim3(VV / 256, MM / 64), 256, 0, stream>>>(
        hbuf, embb, nullptr, (float*)d_out, MM, VV, DD);
}
